// CapsNet_50491635532049
// MI455X (gfx1250) — matrix operand order repaired and checked by reading, not compiled, not run on hardware
//
#include <hip/hip_runtime.h>
#include <hip/hip_bf16.h>
#include <math.h>

// ---------------------------------------------------------------------------
// CDNA5 (gfx1250) CapsNet forward.  wave32.
//   * convs:    implicit GEMM, v_wmma_f32_16x16x32_bf16 (f32 accumulate)
//   * routing:  v_wmma_f32_16x16x4_f32 (full precision)
//   * conv2 B panel staged to LDS by the Tensor Data Mover (tensor_load_to_lds)
// ---------------------------------------------------------------------------

typedef __bf16 bf16_t;
typedef __attribute__((ext_vector_type(8)))  __bf16 v8bf;
typedef __attribute__((ext_vector_type(16))) __bf16 v16bf;
typedef __attribute__((ext_vector_type(2)))  float  v2f;
typedef __attribute__((ext_vector_type(8)))  float  v8f;
typedef __attribute__((ext_vector_type(4)))  unsigned int v4u;
typedef __attribute__((ext_vector_type(4)))  int    v4i;
typedef __attribute__((ext_vector_type(8)))  int    v8i;

#define NB    128          // batch
#define NR    2312         // routes = 8*17*17
#define NC    10           // digit caps
#define NO    16           // out ch per cap
#define NI    8            // in ch per cap
#define NKR   (NR*NI)      // 18496 routing K
#define NCO   (NC*NO)      // 160 routing N
#define K1    256          // conv1 K (243 padded)
#define N1    32
#define M1    (NB*41*41)   // 215168
#define K2    2592         // conv2 K = 32*81
#define N2    64
#define M2    (NB*17*17)   // 36992
#define CHUNKS 68          // split-K chunks for s-GEMM (34 routes each)
#define KPC   272          // K per chunk = 34*8

__device__ __forceinline__ v16bf join8(v8bf lo, v8bf hi) {
  return __builtin_shufflevector(lo, hi, 0,1,2,3,4,5,6,7,8,9,10,11,12,13,14,15);
}

// ---- weight prep: transpose conv weights into [n][k] bf16 panels ----------
__global__ __launch_bounds__(256) void prep_weights(
    const float* __restrict__ conv_w, const float* __restrict__ prim_w,
    bf16_t* __restrict__ B1t, bf16_t* __restrict__ B2t) {
  int gid = blockIdx.x * 256 + threadIdx.x;
  if (gid < N1 * K1) {                       // conv1 panel, k=(c,kh,kw), pad->256
    int n = gid >> 8, k = gid & 255;
    float v = 0.f;
    if (k < 243) {
      int c = k / 81, rem = k - c * 81;
      int kh = rem / 9, kw = rem - kh * 9;
      v = conv_w[((n * 3 + c) * 9 + kh) * 9 + kw];
    }
    B1t[n * K1 + k] = (bf16_t)v;
  }
  int g2 = gid - N1 * K1;
  if (g2 >= 0 && g2 < N2 * K2) {             // conv2 panel, k=(kh,kw,ci)
    int n = g2 / K2, k = g2 - n * K2;
    int khkw = k >> 5, ci = k & 31;
    int kh = khkw / 9, kw = khkw - kh * 9;
    B2t[n * K2 + k] = (bf16_t)prim_w[((n * 32 + ci) * 9 + kh) * 9 + kw];
  }
}

__global__ __launch_bounds__(256) void zero_f32(float* p, int n) {
  int gid = blockIdx.x * 256 + threadIdx.x;
  if (gid < n) p[gid] = 0.f;
}

// ---- conv1: implicit GEMM, im2col staged via LDS, bf16 WMMA ---------------
// block = 128 thr (4 waves); block tile 64M x 32N; wave tile 16M x 32N.
__global__ __launch_bounds__(128) void conv1_wmma(
    const float* __restrict__ data, const float* __restrict__ conv_b,
    const bf16_t* __restrict__ B1t, bf16_t* __restrict__ x1) {
  __shared__ bf16_t Alds[64 * 32];
  const int tid = threadIdx.x;
  const int wave = tid >> 5, lane = tid & 31;
  const int lo16 = lane & 15, hi = lane >> 4;
  const int m0 = blockIdx.x * 64;

  // fill-loop invariants: each thread owns fixed (ml, kk) slots
  const int kk = tid & 31;                  // k within a 32-wide k-step
  const int wq = tid >> 5;
  int rowbase[16], ldsoff[16];
#pragma unroll
  for (int q = 0; q < 16; ++q) {
    int ml = q * 4 + wq;
    int m = m0 + ml;
    int b = m / 1681, p = m - b * 1681;
    int oh = p / 41, ow = p - oh * 41;
    rowbase[q] = b * 7203 + oh * 49 + ow;   // 7203 = 3*49*49
    ldsoff[q] = ml * 32 + kk;
  }

  v8f acc0 = {}; v8f acc1 = {};
  for (int ks = 0; ks < 8; ++ks) {
    const int kb = ks * 32;
    const int k = kb + kk;
    const bool valid = (k < 243);
    int coff = 0;
    if (valid) {
      int c = k / 81, rem = k - c * 81;
      int kh = rem / 9, kw = rem - kh * 9;
      coff = c * 2401 + kh * 49 + kw;       // 2401 = 49*49
    }
    __syncthreads();
#pragma unroll
    for (int q = 0; q < 16; ++q) {
      float val = valid ? data[rowbase[q] + coff] : 0.f;
      Alds[ldsoff[q]] = (bf16_t)val;
    }
    __syncthreads();
    // A fragment: two contiguous 8-elem K runs per lane (ds_load_b128 x2)
    int arow = wave * 16 + lo16;
    v8bf alo = *(const v8bf*)(&Alds[arow * 32 + 16 * hi]);
    v8bf ahi = *(const v8bf*)(&Alds[arow * 32 + 16 * hi + 8]);
    v16bf afrag = join8(alo, ahi);
    // B fragments: contiguous 32B per lane from [n][k] panel (L2 hot)
    v16bf b0 = *(const v16bf*)(&B1t[(0 * 16 + lo16) * K1 + kb + 16 * hi]);
    v16bf b1 = *(const v16bf*)(&B1t[(1 * 16 + lo16) * K1 + kb + 16 * hi]);
    acc0 = __builtin_amdgcn_wmma_f32_16x16x32_bf16(false, afrag, false, b0,
                                                   (short)0, acc0, false, false);
    acc1 = __builtin_amdgcn_wmma_f32_16x16x32_bf16(false, afrag, false, b1,
                                                   (short)0, acc1, false, false);
  }
  // epilogue: bias + ReLU, store NHWC bf16 (feeds conv2 A-fragments)
  float cb0 = conv_b[lo16];
  float cb1 = conv_b[16 + lo16];
#pragma unroll
  for (int j = 0; j < 8; ++j) {
    int m = m0 + wave * 16 + j + 8 * hi;     // C rows: j + 8*hi, col = lo16
    float y0 = fmaxf(acc0[j] + cb0, 0.f);
    float y1 = fmaxf(acc1[j] + cb1, 0.f);
    x1[m * 32 + lo16]      = (bf16_t)y0;
    x1[m * 32 + 16 + lo16] = (bf16_t)y1;
  }
}

// ---- conv2 (primary caps): implicit GEMM, B panel staged via TDM ----------
// block = 128 thr (4 waves); wave tile 16M x 64N (4 accumulators).
// Per k-step, wave 0 issues one tensor_load_to_lds moving the 64x32 bf16
// weight tile (2-D D#) into LDS; all 4 waves consume it (x4 reuse, no VMEM).
__global__ __launch_bounds__(128) void conv2_wmma(
    const bf16_t* __restrict__ x1, const bf16_t* __restrict__ B2t,
    const float* __restrict__ prim_b, float* __restrict__ Uraw) {
  __shared__ bf16_t Blds[64 * 32];
  const int tid = threadIdx.x;
  const int wave = tid >> 5, lane = tid & 31;
  const int lo16 = lane & 15, hi = lane >> 4;
  const int mbase = blockIdx.x * 64 + wave * 16;
  const int mrow = mbase + lo16;             // A row for this lane
  const int b = mrow / 289, p = mrow - b * 289;
  const int oh = p / 17, ow = p - oh * 17;
  const bf16_t* abase = x1 + (((b * 41) + oh * 2) * 41 + ow * 2) * 32;

  const unsigned int ldsa = (unsigned int)(size_t)(&Blds[0]);
  const unsigned long long gbase = (unsigned long long)(size_t)B2t;

  v8f acc[4] = {{}, {}, {}, {}};
  for (int khkw = 0; khkw < 81; ++khkw) {
    __syncthreads();                         // previous tile fully consumed
    if (wave == 0) {
      // ---- Tensor DMA descriptor (D#), 2-D tile: 32 elem x 64 rows, bf16 --
      unsigned long long ga = gbase + (unsigned long long)khkw * 64ull;
      v4u g0 = { 1u,                              // count=1, user descriptor
                 ldsa,                            // lds_addr
                 (unsigned int)ga,                // global_addr[31:0]
                 (unsigned int)(ga >> 32) | (2u << 30) };  // addr[56:32]|type=2
      v8i g1 = { (int)0x10000,                    // data_size=2B, no multicast
                 (int)(2592u << 16),              // tensor_dim0 = 2592 elem
                 (int)(64u << 16),                // tensor_dim1 = 64 rows
                 (int)(32u << 16),                // tile_dim0   = 32 elem
                 (int)64,                         // tile_dim1   = 64 rows
                 (int)2592,                       // tensor_dim0_stride = 2592
                 0, 0 };
      v4i gz4 = { 0, 0, 0, 0 };                   // groups 2/3: unused (2-D)
      v8i gz8 = { 0, 0, 0, 0, 0, 0, 0, 0 };       // extra group (clang-23 form)
      __builtin_amdgcn_tensor_load_to_lds(g0, g1, gz4, gz4, gz8, 0);
      __builtin_amdgcn_s_wait_tensorcnt(0);
    }
    __syncthreads();                         // LDS tile visible to all waves

    int kh = khkw / 9, kw = khkw - kh * 9;
    const bf16_t* ap = abase + (kh * 41 + kw) * 32;
    if (kw == 0) __builtin_prefetch(ap + 41 * 32, 0, 1);   // next input row
    v8bf alo = *(const v8bf*)(ap + 16 * hi);
    v8bf ahi = *(const v8bf*)(ap + 16 * hi + 8);
    v16bf afrag = join8(alo, ahi);
#pragma unroll
    for (int nt = 0; nt < 4; ++nt) {
      v16bf bfrag = *(const v16bf*)(&Blds[(nt * 16 + lo16) * 32 + 16 * hi]);
      acc[nt] = __builtin_amdgcn_wmma_f32_16x16x32_bf16(false, afrag, false, bfrag,
                                                        (short)0, acc[nt], false, false);
    }
  }
  // epilogue: + bias, scatter into Uraw[b][i][r], r = g*289 + oh*17 + ow
#pragma unroll
  for (int nt = 0; nt < 4; ++nt) {
    int n = nt * 16 + lo16;
    int ii = n >> 3, g = n & 7;
    float pb = prim_b[n];
#pragma unroll
    for (int j = 0; j < 8; ++j) {
      int mr = mbase + j + 8 * hi;
      int bb = mr / 289, pp = mr - bb * 289;
      Uraw[(bb * 8 + ii) * NR + (g * 289 + pp)] = acc[nt][j] + pb;
    }
  }
}

// ---- squash over routes: per-(b,i) norm, then scale + build X and X^T -----
__global__ __launch_bounds__(256) void squash_sn(
    const float* __restrict__ Uraw, float* __restrict__ scale) {
  int bi = blockIdx.x;                       // b*8 + i  (1024 blocks)
  const float* row = Uraw + (size_t)bi * NR;
  float s = 0.f;
  for (int r = threadIdx.x; r < NR; r += 256) { float v = row[r]; s += v * v; }
  __shared__ float red[256];
  red[threadIdx.x] = s; __syncthreads();
  for (int st = 128; st > 0; st >>= 1) {
    if (threadIdx.x < st) red[threadIdx.x] += red[threadIdx.x + st];
    __syncthreads();
  }
  if (threadIdx.x == 0) {
    float sn = red[0];
    scale[bi] = sn / ((1.f + sn) * sqrtf(sn));
  }
}

__global__ __launch_bounds__(256) void squash_scatter(
    const float* __restrict__ Uraw, const float* __restrict__ scale,
    float* __restrict__ Xflat, float* __restrict__ Xt) {
  int gid = blockIdx.x * 256 + threadIdx.x;  // over 128*8*2312
  int bi = gid / NR, r = gid - bi * NR;
  int b = bi >> 3, i = bi & 7;
  float v = Uraw[gid] * scale[bi];
  Xflat[((size_t)b * NR + r) * NI + i] = v;  // [b][r*8+i]  (K-contig rows)
  Xt[((size_t)r * NI + i) * NB + b] = v;     // [(r,i)][b]  (K-contig rows)
}

// ---- routing: softmax of b_ij over capsules -------------------------------
__global__ __launch_bounds__(256) void softmax_c(
    const float* __restrict__ bij, float* __restrict__ crc) {
  int r = blockIdx.x * 256 + threadIdx.x;
  if (r >= NR) return;
  float mx = -1e30f;
  float e[NC];
#pragma unroll
  for (int c = 0; c < NC; ++c) mx = fmaxf(mx, bij[r * NC + c]);
  float sum = 0.f;
#pragma unroll
  for (int c = 0; c < NC; ++c) { e[c] = expf(bij[r * NC + c] - mx); sum += e[c]; }
  float inv = 1.f / sum;
#pragma unroll
  for (int c = 0; c < NC; ++c) crc[r * NC + c] = e[c] * inv;
}

// ---- s-GEMM: S[b,(c,o)] = sum_{r,i} (c[r,c]*W[r,c,o,i]) * x[b,r,i] --------
// split-K over 68 chunks (34 routes each); 8 waves, wave = one 16-row M tile,
// 10 accumulators (one per capsule).  Deterministic: partials + fixed-order reduce.
__global__ __launch_bounds__(256) void routing_s_gemm(
    const float* __restrict__ Xflat, const float* __restrict__ W,
    const float* __restrict__ crc, float* __restrict__ Spart) {
  const int tid = threadIdx.x;
  const int wave = tid >> 5, lane = tid & 31;
  const int lo16 = lane & 15, hi = lane >> 4;
  const int chunk = blockIdx.x;
  const int k0 = chunk * KPC;
  v8f acc[NC] = {{}, {}, {}, {}, {}, {}, {}, {}, {}, {}};
  const float* arow = Xflat + (size_t)(wave * 16 + lo16) * NKR;
  for (int t = 0; t < KPC / 4; ++t) {
    int kg = k0 + 4 * t;
    v2f a = *(const v2f*)(arow + kg + 2 * hi);   // A: K pair {2hi,2hi+1}
    int r = kg >> 3;
    int i0 = (kg & 7) + 2 * hi;
    if ((t & 3) == 0)                            // stream hint on W_digit
      __builtin_prefetch(W + (((size_t)(r + 2) * NC) * NO + lo16) * NI, 0, 1);
#pragma unroll
    for (int c = 0; c < NC; ++c) {
      float cs = crc[r * NC + c];                // fuse softmax coeff into B
      v2f wv = *(const v2f*)(W + ((((size_t)r * NC + c) * NO + lo16) * NI + i0));
      v2f bf = wv * cs;
      acc[c] = __builtin_amdgcn_wmma_f32_16x16x4_f32(false, a, false, bf,
                                                     (short)0, acc[c], false, false);
    }
  }
  float* sp = Spart + (size_t)chunk * NB * NCO;
#pragma unroll
  for (int c = 0; c < NC; ++c)
#pragma unroll
    for (int j = 0; j < 8; ++j) {
      int brow = wave * 16 + j + 8 * hi;
      sp[brow * NCO + c * NO + lo16] = acc[c][j];
    }
}

// ---- reduce split-K partials + squash(v) ; also emit V^T for a-GEMM -------
__global__ __launch_bounds__(128) void reduce_squash_v(
    const float* __restrict__ Spart, float* __restrict__ V, float* __restrict__ Vt) {
  const int tid = threadIdx.x;
  const int wave = tid >> 5, lane = tid & 31;
  const int wg = blockIdx.x * 4 + wave;        // 0..1279 = b*10 + c
  const int b = wg / NC, c = wg - b * NC;
  if (lane < 16) {
    int o = lane;
    float s = 0.f;
    for (int ch = 0; ch < CHUNKS; ++ch)        // fixed order: deterministic
      s += Spart[((size_t)ch * NB + b) * NCO + c * NO + o];
    float ss = s * s;
    for (int mask = 1; mask < 16; mask <<= 1) ss += __shfl_xor(ss, mask, 32);
    float sn = ss;
    float v = s * sn / ((1.f + sn) * sqrtf(sn));
    V[b * NCO + c * NO + o] = v;
    Vt[(c * NO + o) * NB + b] = v;
  }
}

// ---- a-GEMM: T = X^T (18496x128) * V (128x160), contracted with W in-reg ---
// block = 4 waves; wave = one 16-row tile (2 routes) x all 160 cols.
__global__ __launch_bounds__(128) void routing_a_gemm(
    const float* __restrict__ Xt, const float* __restrict__ Vt,
    const float* __restrict__ W, float* __restrict__ bij) {
  const int tid = threadIdx.x;
  const int wave = tid >> 5, lane = tid & 31;
  const int lo16 = lane & 15, hi = lane >> 4;
  const int mrow0 = blockIdx.x * 64 + wave * 16;
  v8f acc[NC] = {{}, {}, {}, {}, {}, {}, {}, {}, {}, {}};
  const float* arow = Xt + (size_t)(mrow0 + lo16) * NB;
  for (int t = 0; t < NB / 4; ++t) {
    int kg = 4 * t;
    v2f a = *(const v2f*)(arow + kg + 2 * hi);
#pragma unroll
    for (int c = 0; c < NC; ++c) {
      v2f bf = *(const v2f*)(Vt + (c * NO + lo16) * NB + kg + 2 * hi);
      acc[c] = __builtin_amdgcn_wmma_f32_16x16x4_f32(false, a, false, bf,
                                                     (short)0, acc[c], false, false);
    }
  }
  // a[r,c] = (1/B) * sum_{i,o} W[r,c,o,i] * T[(r,i),(c,o)]
  // C rows j+8*hi -> route = hi half, i = j ; col = o = lo16
  const int rg = blockIdx.x * 8 + wave * 2 + hi;
#pragma unroll
  for (int c = 0; c < NC; ++c) {
    const float* wp = W + (((size_t)rg * NC + c) * NO + lo16) * NI;
    float part = 0.f;
#pragma unroll
    for (int j = 0; j < 8; ++j) part += acc[c][j] * wp[j];
    for (int mask = 1; mask < 16; mask <<= 1) part += __shfl_xor(part, mask, 32);
    if (lo16 == 0)                               // unique owner -> plain RMW
      bij[rg * NC + c] += part * (1.f / 128.f);
  }
}

// ---- classifier head + emit outputs ---------------------------------------
__global__ __launch_bounds__(256) void head_kernel(
    const float* __restrict__ V, const float* __restrict__ w1,
    const float* __restrict__ b1, const float* __restrict__ w2,
    const float* __restrict__ b2, float* __restrict__ out) {
  const int tid = threadIdx.x;
  const int wave = tid >> 5, lane = tid & 31;
  const int wg = blockIdx.x * 8 + wave;          // 0..1279 = b*10 + c
  const int b = wg / NC, c = wg - b * NC;
  const float* vp = V + b * NCO + c * NO;
  if (lane < 16) out[(b * NC + c) * NO + lane] = vp[lane];   // v -> output[0]
  float vloc[NO];
#pragma unroll
  for (int i = 0; i < NO; ++i) vloc[i] = vp[i];
  float part = 0.f;
  for (int h = lane; h < 100; h += 32) {
    float d = b1[c * 100 + h];
#pragma unroll
    for (int i = 0; i < NO; ++i) d += vloc[i] * w1[(c * NO + i) * 100 + h];
    part += tanhf(d) * w2[c * 100 + h];
  }
  for (int mask = 1; mask < 32; mask <<= 1) part += __shfl_xor(part, mask, 32);
  if (lane == 0) {
    float logit = part + b2[c];
    out[20480 + c * NB + b] = 1.f / (1.f + expf(-logit));
  }
}

// ---------------------------------------------------------------------------
extern "C" void kernel_launch(void* const* d_in, const int* in_sizes, int n_in,
                              void* d_out, int out_size, void* d_ws, size_t ws_size,
                              hipStream_t stream) {
  const float* data   = (const float*)d_in[0];
  const float* conv_w = (const float*)d_in[1];
  const float* conv_b = (const float*)d_in[2];
  const float* prim_w = (const float*)d_in[3];
  const float* prim_b = (const float*)d_in[4];
  const float* Wdig   = (const float*)d_in[5];
  const float* h_w1   = (const float*)d_in[6];
  const float* h_b1   = (const float*)d_in[7];
  const float* h_w2   = (const float*)d_in[8];
  const float* h_b2   = (const float*)d_in[9];
  float* out = (float*)d_out;
  (void)in_sizes; (void)n_in; (void)out_size; (void)ws_size;

  char* base = (char*)d_ws;
  size_t off = 0;
  auto take = [&](size_t bytes) -> char* {
    char* p = base + off;
    off = (off + bytes + 255) & ~(size_t)255;
    return p;
  };
  bf16_t* B1t  = (bf16_t*)take((size_t)N1 * K1 * 2);          //  16 KB
  bf16_t* B2t  = (bf16_t*)take((size_t)N2 * K2 * 2);          // 324 KB
  bf16_t* X1   = (bf16_t*)take((size_t)M1 * 32 * 2);          // 13.1 MB, NHWC
  float*  Uraw = (float*)take((size_t)NB * 8 * NR * 4);       // 9.5 MB
  float*  Scal = (float*)take((size_t)NB * 8 * 4);
  float*  Xfl  = (float*)take((size_t)NB * NR * NI * 4);      // 9.5 MB
  float*  Xt   = (float*)take((size_t)NR * NI * NB * 4);      // 9.5 MB
  float*  Bij  = (float*)take((size_t)NR * NC * 4);
  float*  Crc  = (float*)take((size_t)NR * NC * 4);
  float*  Spt  = (float*)take((size_t)CHUNKS * NB * NCO * 4); // 5.6 MB
  float*  V    = (float*)take((size_t)NB * NCO * 4);
  float*  Vt   = (float*)take((size_t)NCO * NB * 4);

  prep_weights<<<(N1 * K1 + N2 * K2 + 255) / 256, 256, 0, stream>>>(conv_w, prim_w, B1t, B2t);
  zero_f32<<<(NR * NC + 255) / 256, 256, 0, stream>>>(Bij, NR * NC);
  conv1_wmma<<<M1 / 64, 128, 0, stream>>>(data, conv_b, B1t, X1);
  conv2_wmma<<<M2 / 64, 128, 0, stream>>>(X1, B2t, prim_b, Uraw);
  squash_sn<<<NB * 8, 256, 0, stream>>>(Uraw, Scal);
  squash_scatter<<<(NB * 8 * NR) / 256, 256, 0, stream>>>(Uraw, Scal, Xfl, Xt);

  for (int it = 0; it < 3; ++it) {
    softmax_c<<<(NR + 255) / 256, 256, 0, stream>>>(Bij, Crc);
    routing_s_gemm<<<CHUNKS, 256, 0, stream>>>(Xfl, Wdig, Crc, Spt);
    reduce_squash_v<<<(NB * NC) / 4, 128, 0, stream>>>(Spt, V, Vt);
    if (it < 2)
      routing_a_gemm<<<NR / 8, 128, 0, stream>>>(Xt, Vt, Wdig, Bij);
  }
  head_kernel<<<(NB * NC) / 8, 256, 0, stream>>>(V, h_w1, h_b1, h_w2, h_b2, out);
}
